// NTXentLoss_19722489823618
// MI455X (gfx1250) — matrix operand order repaired and checked by reading, not compiled, not run on hardware
//
#include <hip/hip_runtime.h>

typedef _Float16 v8h   __attribute__((ext_vector_type(8)));
typedef _Float16 v16h  __attribute__((ext_vector_type(16)));
typedef float    v8f   __attribute__((ext_vector_type(8)));
typedef float    vf4   __attribute__((ext_vector_type(4)));

#define NROWS      8192           // 2N
#define HALFN      4096           // N
#define DIMK       256            // D
#define ROWTILES   512            // NROWS/16
#define COLCHUNKS  4              // column-space split
#define TILES_PER_CHUNK 128       // ROWTILES/COLCHUNKS
#define WAVES_PER_BLOCK 8
// exp(s/T) computed as exp2(s * log2(e)/T)
#define SCALE_EXP2 (1.4426950408889634f / 0.07f)
#define INV_T      (1.0f / 0.07f)
#define LN2F       0.6931471805599453f
// LDS row stride in halfs: 256 data + 8 pad -> 528B stride = 132 dwords,
// lane m starts at bank (4m mod 64): conflict-free per 16-lane half.
#define LDS_STRIDE_H 264

// sched_group_barrier masks
#define SGB_MFMA_WMMA 0x008
#define SGB_DS_READ   0x100

// ---------------------------------------------------------------------------
// Kernel 1: L2-normalize each of the 8192 rows; emit f16 matrix + 1/norm.
// ---------------------------------------------------------------------------
__global__ void __launch_bounds__(256)
ntx_normalize(const float* __restrict__ zi, const float* __restrict__ zj,
              _Float16* __restrict__ Ah, float* __restrict__ invnorm) {
  const int wave = (blockIdx.x * blockDim.x + threadIdx.x) >> 5;
  const int lane = threadIdx.x & 31;
  if (wave >= NROWS) return;
  const float* src = (wave < HALFN) ? (zi + (size_t)wave * DIMK)
                                    : (zj + (size_t)(wave - HALFN) * DIMK);
  vf4 x0 = *(const vf4*)(src + lane * 8);
  vf4 x1 = *(const vf4*)(src + lane * 8 + 4);
  float ss = 0.f;
#pragma unroll
  for (int k = 0; k < 4; ++k) ss += x0[k] * x0[k] + x1[k] * x1[k];
#pragma unroll
  for (int m = 16; m >= 1; m >>= 1) ss += __shfl_xor(ss, m, 32);
  const float inv = 1.0f / fmaxf(sqrtf(ss), 1e-12f);
  v8h h;
#pragma unroll
  for (int k = 0; k < 4; ++k) {
    h[k]     = (_Float16)(x0[k] * inv);
    h[4 + k] = (_Float16)(x1[k] * inv);
  }
  *(v8h*)(Ah + (size_t)wave * DIMK + lane * 8) = h;
  if (lane == 0) invnorm[wave] = inv;
}

// ---------------------------------------------------------------------------
// Kernel 2: exact fp32 positive-pair logits s_pos[i] = <zn_i, zn_{i+N}> / T.
// ---------------------------------------------------------------------------
__global__ void __launch_bounds__(256)
ntx_posdot(const float* __restrict__ zi, const float* __restrict__ zj,
           const float* __restrict__ invnorm, float* __restrict__ poss) {
  const int wave = (blockIdx.x * blockDim.x + threadIdx.x) >> 5;
  const int lane = threadIdx.x & 31;
  if (wave >= HALFN) return;
  const float* a = zi + (size_t)wave * DIMK;
  const float* b = zj + (size_t)wave * DIMK;
  vf4 a0 = *(const vf4*)(a + lane * 8);
  vf4 a1 = *(const vf4*)(a + lane * 8 + 4);
  vf4 b0 = *(const vf4*)(b + lane * 8);
  vf4 b1 = *(const vf4*)(b + lane * 8 + 4);
  float d = 0.f;
#pragma unroll
  for (int k = 0; k < 4; ++k) d += a0[k] * b0[k] + a1[k] * b1[k];
#pragma unroll
  for (int m = 16; m >= 1; m >>= 1) d += __shfl_xor(d, m, 32);
  if (lane == 0) {
    const float s = d * invnorm[wave] * invnorm[wave + HALFN] * INV_T;
    poss[wave] = s;
    poss[wave + HALFN] = s;
  }
}

// ---------------------------------------------------------------------------
// Async global->LDS staging of one 16-row column tile (8 KB).
// 512 x 16B quanta; 256 threads issue 2 async copies each (ASYNCcnt-tracked
// per-lane DMA). Low 32 bits of a flat __shared__ pointer are the LDS-base-
// relative byte address (ISA Sec 10.2: flat LDS addresses are truncated).
// ---------------------------------------------------------------------------
__device__ __forceinline__ void stage_tile(const _Float16* __restrict__ Ah,
                                           int jt, _Float16* lb) {
#pragma unroll
  for (int k = 0; k < 2; ++k) {
    const int q   = (int)threadIdx.x + k * 256;  // 16-byte quantum id 0..511
    const int row = q >> 5;                      // 0..15
    const int c16 = q & 31;                      // 16B column within row
    const _Float16* g = Ah + (size_t)(jt * 16 + row) * DIMK + c16 * 8;
    const unsigned lds = (unsigned)(size_t)(lb + row * LDS_STRIDE_H + c16 * 8);
    asm volatile("global_load_async_to_lds_b128 %0, %1, off"
                 :: "v"(lds), "v"(g)
                 : "memory");
  }
}

__device__ __forceinline__ void wait_async_lds() {
  asm volatile("s_wait_asynccnt 0x0" ::: "memory");
}

// ---------------------------------------------------------------------------
// Kernel 3: streaming WMMA sim-row-sum with LDS-staged, double-buffered
// column tiles shared by all 8 waves of the block.
//   block -> (row block rb of 8 row tiles) x (column chunk of 128 tiles)
//   wave w -> row tile r = rb*8+w ; all waves sweep the same column tiles.
// sched_group_barrier pins each tile body to [16x ds_read][8x wmma] so all
// eight B chunks stay live and the WMMAs run back-to-back behind one wait.
// ---------------------------------------------------------------------------
__global__ void __launch_bounds__(256)
__attribute__((amdgpu_waves_per_eu(1, 2)))
ntx_simsum(const _Float16* __restrict__ Ah, float* __restrict__ partial) {
  __shared__ _Float16 lbuf[2][16 * LDS_STRIDE_H];

  const int lane  = threadIdx.x & 31;
  const int wv    = threadIdx.x >> 5;       // wave in block 0..7
  const int rb    = blockIdx.x >> 2;        // row block 0..63
  const int chunk = blockIdx.x & 3;         // column chunk 0..3
  const int r     = rb * WAVES_PER_BLOCK + wv;  // row tile 0..511
  const int m  = lane & 15;
  const int hi = lane >> 4;

  // A tile: rows r*16+m, contiguous K order (lanes 0-15: K=[0..15];
  // lanes 16-31: K=[16..31] per 32-wide K chunk). Held for whole sweep.
  const _Float16* arow = Ah + (size_t)(r * 16 + m) * DIMK;
  v16h Areg[8];
#pragma unroll
  for (int c = 0; c < 8; ++c) {
    Areg[c] = *(const v16h*)(arow + c * 32 + hi * 16);
  }

  const int jt0 = chunk * TILES_PER_CHUNK;
  const int jt1 = jt0 + TILES_PER_CHUNK;

  // Prologue: stage first tile, wait for this wave's DMA, barrier.
  stage_tile(Ah, jt0, &lbuf[0][0]);
  wait_async_lds();
  __syncthreads();

  v8f rs = {0.f, 0.f, 0.f, 0.f, 0.f, 0.f, 0.f, 0.f};
  for (int jt = jt0; jt < jt1; ++jt) {
    const int p = (jt - jt0) & 1;
    // Kick off next tile's global->LDS DMA; overlaps with this tile's WMMAs.
    if (jt + 1 < jt1) stage_tile(Ah, jt + 1, &lbuf[p ^ 1][0]);

    // B tile (Zn^T) from LDS: lane = column N = row jt*16+m; 16 contiguous
    // halfs per K-chunk.
    const _Float16* brow = &lbuf[p][m * LDS_STRIDE_H + hi * 16];
    v16h B[8];
#pragma unroll
    for (int c = 0; c < 8; ++c) B[c] = *(const v16h*)(brow + c * 32);

    v8f acc0 = {0.f, 0.f, 0.f, 0.f, 0.f, 0.f, 0.f, 0.f};
    v8f acc1 = {0.f, 0.f, 0.f, 0.f, 0.f, 0.f, 0.f, 0.f};
#pragma unroll
    for (int c = 0; c < 8; c += 2) {
      acc0 = __builtin_amdgcn_wmma_f32_16x16x32_f16(false, Areg[c],     false, B[c],
                                                    (short)0, acc0, false, false);
      acc1 = __builtin_amdgcn_wmma_f32_16x16x32_f16(false, Areg[c + 1], false, B[c + 1],
                                                    (short)0, acc1, false, false);
    }
    // Force schedule: all 16 ds_load_b128 first (one clause + one wait),
    // then the 8 WMMAs back-to-back.
    __builtin_amdgcn_sched_group_barrier(SGB_DS_READ,   16, 0);
    __builtin_amdgcn_sched_group_barrier(SGB_MFMA_WMMA,  8, 0);

    v8f acc = acc0 + acc1;   // cosine sims in [-1,1]

    if (jt == r) {
      // diagonal tile: drop self-similarity (lane v and lane 24+v in VGPR v)
#pragma unroll
      for (int v = 0; v < 8; ++v) {
        float e = __builtin_amdgcn_exp2f(acc[v] * SCALE_EXP2);
        if (lane == v || lane == 24 + v) e = 0.f;
        rs[v] += e;
      }
    } else {
#pragma unroll
      for (int v = 0; v < 8; ++v)
        rs[v] += __builtin_amdgcn_exp2f(acc[v] * SCALE_EXP2);
    }

    // Our DMA for the next buffer must have landed before the barrier; the
    // barrier then guarantees everyone is done reading buf[p] and writing
    // buf[p^1].
    wait_async_lds();
    __syncthreads();
  }

  // Reduce over the 16 N-columns held across each 16-lane half.
#pragma unroll
  for (int v = 0; v < 8; ++v) {
    float s = rs[v];
    s += __shfl_xor(s, 1, 32);
    s += __shfl_xor(s, 2, 32);
    s += __shfl_xor(s, 4, 32);
    s += __shfl_xor(s, 8, 32);
    // lanes 0-15 hold sum for row r*16+v; lanes 16-31 for row r*16+8+v
    if (lane == 0)
      partial[(size_t)chunk * NROWS + r * 16 + v] = s;
    if (lane == 16)
      partial[(size_t)chunk * NROWS + r * 16 + 8 + v] = s;
  }
}

// ---------------------------------------------------------------------------
// Kernel 4: loss = mean_i( log(rowsum_i) - s_pos_i ), fixed-order reduction.
// ---------------------------------------------------------------------------
__global__ void __launch_bounds__(256)
ntx_loss(const float* __restrict__ partial, const float* __restrict__ poss,
         float* __restrict__ out) {
  __shared__ float sm[256];
  float local = 0.f;
  for (int i = threadIdx.x; i < NROWS; i += 256) {
    const float rsum = partial[i] + partial[NROWS + i] +
                       partial[2 * NROWS + i] + partial[3 * NROWS + i];
    local += __builtin_amdgcn_logf(rsum) * LN2F - poss[i];
  }
  sm[threadIdx.x] = local;
  __syncthreads();
  for (int s = 128; s > 0; s >>= 1) {
    if ((int)threadIdx.x < s) sm[threadIdx.x] += sm[threadIdx.x + s];
    __syncthreads();
  }
  if (threadIdx.x == 0) out[0] = sm[0] / (float)NROWS;
}

// ---------------------------------------------------------------------------
extern "C" void kernel_launch(void* const* d_in, const int* in_sizes, int n_in,
                              void* d_out, int out_size, void* d_ws, size_t ws_size,
                              hipStream_t stream) {
  (void)in_sizes; (void)n_in; (void)out_size; (void)ws_size;
  const float* zi = (const float*)d_in[0];
  const float* zj = (const float*)d_in[1];
  float* out = (float*)d_out;

  char* ws = (char*)d_ws;
  _Float16* Ah      = (_Float16*)ws;                                   // 4 MB
  float*    invnorm = (float*)(ws + (size_t)NROWS * DIMK * 2);         // 32 KB
  float*    poss    = (float*)(ws + (size_t)NROWS * DIMK * 2 + 32768); // 32 KB
  float*    partial = (float*)(ws + (size_t)NROWS * DIMK * 2 + 65536); // 128 KB

  ntx_normalize<<<NROWS / 8, 256, 0, stream>>>(zi, zj, Ah, invnorm);
  ntx_posdot<<<HALFN / 8, 256, 0, stream>>>(zi, zj, invnorm, poss);
  // 64 row blocks x 4 column chunks = 256 blocks of 8 waves
  ntx_simsum<<<(ROWTILES / WAVES_PER_BLOCK) * COLCHUNKS, 256, 0, stream>>>(Ah, partial);
  ntx_loss<<<1, 256, 0, stream>>>(partial, poss, out);
}
